// FLD_53644141527684
// MI455X (gfx1250) — compile-verified, matrix operand order checked by reading, not run on hardware
//
#include <hip/hip_runtime.h>
#include <hip/hip_bf16.h>
#include <math.h>

// ---------------- problem constants ----------------
#define BB   32
#define LL   2048
#define DD   64
#define TT   512
#define HH   2
#define EE   128
#define PP   2
#define LAT  64
#define HID  64
#define DIM2 128

#define SPLIT  16                    // L split for k_attn parallelism
#define LSEG   (LL/SPLIT)            // 128 rows per block
#define CH     32                    // rows per chunk (= WMMA K)
#define CHUNKS (LSEG/CH)             // 4

#define OUT0 (BB*TT*DD)              // floats of first tuple element (out)

// workspace layout (floats)
#define WS_A      0                          // 128*4
#define WS_CC     512                        // 4
#define WS_SCORES 1024                       // B*4*L = 262144 (expw in place)
#define WS_XATT   (1024 + BB*4*LL)           // B*SPLIT*4*128 = 262144 partials

typedef __attribute__((ext_vector_type(16))) _Float16 v16h;
typedef __attribute__((ext_vector_type(8)))  _Float16 v8h;
typedef __attribute__((ext_vector_type(8)))  float    v8f;

union H4 { _Float16 h[4]; uint2 u; };

#define SHUF16(lo, hi) __builtin_shufflevector(lo, hi, 0,1,2,3,4,5,6,7,8,9,10,11,12,13,14,15)

// ---------------------------------------------------------------------------
// K1: q = query@qm_w+qm_b (2x128);  A[j,hp] = (1/8)*sum_e km_w[j,h*64+e]*q[p,.]
// ---------------------------------------------------------------------------
__global__ void k_prep(const float* __restrict__ query,
                       const float* __restrict__ qm_w, const float* __restrict__ qm_b,
                       const float* __restrict__ km_w, const float* __restrict__ km_b,
                       float* __restrict__ A, float* __restrict__ cvec) {
  __shared__ float q[2][EE];
  int e = threadIdx.x;                       // 128 threads
  for (int p = 0; p < PP; ++p) {
    float acc = qm_b[e];
    for (int j = 0; j < EE; ++j) acc += query[p*EE + j] * qm_w[j*EE + e];
    q[p][e] = acc;
  }
  __syncthreads();
  int j = threadIdx.x;
  for (int hp = 0; hp < 4; ++hp) {
    int h = hp >> 1, p = hp & 1;
    float acc = 0.f;
    for (int ee = 0; ee < 64; ++ee) acc += km_w[j*EE + h*64 + ee] * q[p][h*64 + ee];
    A[j*4 + hp] = acc * 0.125f;              // 1/sqrt(EK)
  }
  if (threadIdx.x < 4) {
    int hp = threadIdx.x, h = hp >> 1, p = hp & 1;
    float acc = 0.f;
    for (int ee = 0; ee < 64; ++ee) acc += km_b[h*64 + ee] * q[p][h*64 + ee];
    cvec[hp] = acc * 0.125f;
  }
}

// ---------------------------------------------------------------------------
// K2: scores[b,hp,l] = sum_j emb_j(t) * A[j,hp] + c[hp], emb_j = sin on even j
// ---------------------------------------------------------------------------
__global__ void k_scores(const float* __restrict__ ts,
                         const float* __restrict__ te_w, const float* __restrict__ te_b,
                         const float* __restrict__ A, const float* __restrict__ cvec,
                         float* __restrict__ scores) {
  __shared__ float sA[EE*4];
  __shared__ float swv[EE], sbv[EE];
  __shared__ float sc[4];
  int t = threadIdx.x;                       // 256 threads
  for (int i = t; i < EE*4; i += 256) sA[i] = A[i];
  if (t < EE) { swv[t] = te_w[t]; sbv[t] = te_b[t]; }
  if (t < 4)  sc[t] = cvec[t];
  __syncthreads();

  int idx = blockIdx.x * 256 + t;            // b*L + l
  float tv = ts[idx];
  float a0 = sc[0], a1 = sc[1], a2 = sc[2], a3 = sc[3];
#pragma unroll 8
  for (int j = 0; j < EE; j += 2) {
    float e0 = sinf(tv * swv[j] + sbv[j]);   // even index -> sin
    float e1 =      tv * swv[j+1] + sbv[j+1];
    a0 += e0 * sA[j*4+0] + e1 * sA[(j+1)*4+0];
    a1 += e0 * sA[j*4+1] + e1 * sA[(j+1)*4+1];
    a2 += e0 * sA[j*4+2] + e1 * sA[(j+1)*4+2];
    a3 += e0 * sA[j*4+3] + e1 * sA[(j+1)*4+3];
  }
  int b = idx >> 11, l = idx & (LL-1);
  float* srow = scores + (size_t)b*4*LL + l;
  srow[0*LL] = a0; srow[1*LL] = a1; srow[2*LL] = a2; srow[3*LL] = a3;
}

// ---------------------------------------------------------------------------
// K3: per (b,hp) row: m = max_l s; s <- exp(s-m)  (in place)
// ---------------------------------------------------------------------------
__global__ void k_expw(float* __restrict__ scores) {
  __shared__ float red[256];
  float* s = scores + (size_t)blockIdx.x * LL;
  float m = -3.4e38f;
  for (int i = threadIdx.x; i < LL; i += 256) m = fmaxf(m, s[i]);
  red[threadIdx.x] = m;
  __syncthreads();
  for (int st = 128; st > 0; st >>= 1) {
    if (threadIdx.x < st) red[threadIdx.x] = fmaxf(red[threadIdx.x], red[threadIdx.x + st]);
    __syncthreads();
  }
  m = red[0];
  for (int i = threadIdx.x; i < LL; i += 256) s[i] = expf(s[i] - m);
}

// ---------------------------------------------------------------------------
// K4: per (b, L-segment): WMMA GEMM (16 x 128) expw  x  (128 x 128) [M*X | M].
// A rows 0..3 = hp weight rows; rows 4..15 are hard zeros in LDS (no selects,
// no WMMA->VALU hazards). Values staged transposed f16, packed b64 DS stores.
// grid = B*SPLIT blocks x 256 threads (8 waves; wave nt owns 16 columns).
// ---------------------------------------------------------------------------
#define EWS 136      // sEW row stride (halves): 272B, 16B aligned, 4-bank step
#define VTS 40       // sVt row stride (halves):  80B, 16B aligned, conflict-free
__global__ void k_attn(const float* __restrict__ X, const float* __restrict__ Mm,
                       const float* __restrict__ expw, float* __restrict__ xout) {
  __shared__ __align__(16) _Float16 sEW[16*EWS];     // 16 A rows (12 zero)
  __shared__ __align__(16) _Float16 sVt[DIM2*VTS];   // V chunk, transposed
  int bs = blockIdx.x;
  int b = bs >> 4, s = bs & (SPLIT-1);
  int l0 = s * LSEG;
  int tid = threadIdx.x;

  if (tid < LSEG) {
#pragma unroll
    for (int hp = 0; hp < 4; ++hp)
      sEW[hp*EWS + tid] = (_Float16)expw[((size_t)b*4 + hp)*LL + l0 + tid];
  }
  for (int i = tid; i < 12*EWS; i += 256) sEW[4*EWS + i] = (_Float16)0.0f;

  const float* Xb = X  + ((size_t)b*LL + l0)*DD;
  const float* Mb = Mm + ((size_t)b*LL + l0)*DD;

  int lane = tid & 31;
  int nt   = tid >> 5;                        // wave = column tile 0..7
  int cn   = (nt << 4) + (lane & 15);         // my column 0..127
  int m    = lane & 15;                       // A-matrix row (>=4 read zeros)
  int loff = (lane & 16) ? 8 : 0;
  int sc_  = tid & 63;                        // staging column 0..63
  int rg   = tid >> 6;                        // staging row group 0..3

  v8f acc = {};
  for (int ch = 0; ch < CHUNKS; ++ch) {
    __syncthreads();                          // sVt reuse / sEW ready
    // ---- stage 32 rows of [M*X | M], transposed f16, packed b64 stores ----
#pragma unroll
    for (int j = 0; j < 2; ++j) {
      int lc = rg*4 + j*16;                   // local row base (multiple of 4)
      int l  = ch*CH + lc;
      H4 xm, mm;
#pragma unroll
      for (int r = 0; r < 4; ++r) {
        float xv = Xb[(l + r)*DD + sc_];
        float mv = Mb[(l + r)*DD + sc_];
        xm.h[r] = (_Float16)(xv * mv);
        mm.h[r] = (_Float16)mv;
      }
      *(uint2*)&sVt[sc_*VTS + lc]        = xm.u;
      *(uint2*)&sVt[(64 + sc_)*VTS + lc] = mm.u;
    }
    if (ch + 1 < CHUNKS) {
      __builtin_prefetch(&Xb[(ch*CH + CH + rg*4)*DD + sc_], 0, 1);
      __builtin_prefetch(&Mb[(ch*CH + CH + rg*4)*DD + sc_], 0, 1);
    }
    __syncthreads();
    // ---- fragments: pure aligned b128 LDS loads, no VALU in between ----
    int kb = ch*CH;
    v16h a  = SHUF16(*(const v8h*)&sEW[m*EWS + kb + loff],
                     *(const v8h*)&sEW[m*EWS + kb + loff + 16]);
    v16h bf = SHUF16(*(const v8h*)&sVt[cn*VTS + loff],
                     *(const v8h*)&sVt[cn*VTS + loff + 16]);
    acc = __builtin_amdgcn_wmma_f32_16x16x32_f16(false, a, false, bf,
                                                 (short)0, acc, false, false);
  }
  if (lane < 16) {
#pragma unroll
    for (int r = 0; r < 4; ++r)
      xout[(((size_t)b*SPLIT + s)*4 + r)*DIM2 + (nt << 4) + lane] = acc[r];
  }
}

// ---------------------------------------------------------------------------
// K5: reduce split partials, divide num/den, coeffs = x @ out_w + out_b
// ---------------------------------------------------------------------------
__global__ void k_coeffs(const float* __restrict__ part,
                         const float* __restrict__ out_w, const float* __restrict__ out_b,
                         float* __restrict__ coeffs) {
  __shared__ float sS[4][DIM2];
  __shared__ float xd[4][DIM2];
  int b = blockIdx.x;
  int d = threadIdx.x;                        // 128 threads
#pragma unroll
  for (int hp = 0; hp < 4; ++hp) {
    float acc = 0.f;
    for (int s = 0; s < SPLIT; ++s)
      acc += part[(((size_t)b*SPLIT + s)*4 + hp)*DIM2 + d];
    sS[hp][d] = acc;
  }
  __syncthreads();
#pragma unroll
  for (int hp = 0; hp < 4; ++hp) {
    float den = sS[hp][64 + (d & 63)];
    float num = (d < 64) ? sS[hp][d] : den;
    xd[hp][d] = (den != 0.0f) ? (num / den) : 0.0f;
  }
  __syncthreads();
  if (threadIdx.x < LAT) {
    int lat = threadIdx.x;
    for (int p = 0; p < PP; ++p) {
      float acc = out_b[lat];
      for (int j = 0; j < HH*DIM2; ++j) {
        int h = j >> 7, dd = j & 127;
        acc += xd[h*2 + p][dd] * out_w[j*LAT + lat];
      }
      coeffs[((size_t)b*PP + p)*LAT + lat] = acc;
    }
  }
}

// ---------------------------------------------------------------------------
// K6: decoder MLP per 16-row tile: z = c0 + y*c1 -> relu(z@W1+b1) -> @W2+b2.
// Weights transposed in LDS (packed b64 staging), hidden tile stored f16.
// ---------------------------------------------------------------------------
#define WTS 72       // transposed-weight / h1 row stride (halves): 144B aligned
__global__ void k_decoder(const float* __restrict__ yts, const float* __restrict__ coeffs,
                          const float* __restrict__ w1, const float* __restrict__ b1,
                          const float* __restrict__ w2, const float* __restrict__ b2,
                          float* __restrict__ out) {
  __shared__ __align__(16) _Float16 sw1T[HID*WTS];   // [n][k]
  __shared__ __align__(16) _Float16 sw2T[DD*WTS];    // [n][k]
  __shared__ __align__(16) _Float16 sh1[8][16*WTS];  // per-wave hidden [m][k]
  __shared__ float scoef[BB*PP*LAT];                 // all coeffs, 16 KB
  __shared__ float sb1[HID], sb2[DD];
  int tid = threadIdx.x;                      // 256 threads = 8 waves
  for (int i = tid; i < (HID*LAT)/4; i += 256) {     // packed transposed stage
    int n = i & 63, k4 = (i >> 6) * 4;
    H4 a1v, a2v;
#pragma unroll
    for (int r = 0; r < 4; ++r) {
      a1v.h[r] = (_Float16)w1[(k4 + r)*64 + n];
      a2v.h[r] = (_Float16)w2[(k4 + r)*64 + n];
    }
    *(uint2*)&sw1T[n*WTS + k4] = a1v.u;
    *(uint2*)&sw2T[n*WTS + k4] = a2v.u;
  }
  for (int i = tid; i < BB*PP*LAT; i += 256) scoef[i] = coeffs[i];
  if (tid < HID) { sb1[tid] = b1[tid]; sb2[tid] = b2[tid]; }
  __syncthreads();

  int wave = tid >> 5, lane = tid & 31;
  int gw = blockIdx.x * 8 + wave;             // 0..255 (grid = 32 blocks)
  int m  = lane & 15;
  int n15 = lane & 15;
  int loff = (lane & 16) ? 8 : 0;
  int mb = (lane < 16) ? 0 : 8;
  _Float16* h1 = &sh1[wave][0];

  for (int tile = gw; tile < (BB*TT)/16; tile += 256) {   // uniform trip count
    int row0 = tile * 16;
    int b  = row0 >> 9;                       // / T
    int t0 = row0 & (TT-1);
    float yv = yts[(size_t)b*TT + t0 + m];
    const float* c0 = &scoef[((size_t)b*PP + 0)*LAT];
    const float* c1 = c0 + LAT;

    // z A fragments (contiguous LDS runs -> merged ds loads)
    v16h za0, za1;
#pragma unroll
    for (int i = 0; i < 8; ++i) {
      int k = loff + i;
      za0[i]     = (_Float16)(c0[k]      + yv * c1[k]);
      za0[i + 8] = (_Float16)(c0[k + 16] + yv * c1[k + 16]);
      za1[i]     = (_Float16)(c0[k + 32] + yv * c1[k + 32]);
      za1[i + 8] = (_Float16)(c0[k + 48] + yv * c1[k + 48]);
    }
    // ---- layer 1 ----
#pragma unroll
    for (int nt = 0; nt < 4; ++nt) {
      int n = (nt << 4) + n15;
      v8f acc;
#pragma unroll
      for (int r = 0; r < 8; ++r) acc[r] = sb1[n];
      v16h bw0 = SHUF16(*(const v8h*)&sw1T[n*WTS + loff],
                        *(const v8h*)&sw1T[n*WTS + loff + 16]);
      v16h bw1 = SHUF16(*(const v8h*)&sw1T[n*WTS + 32 + loff],
                        *(const v8h*)&sw1T[n*WTS + 32 + loff + 16]);
      acc = __builtin_amdgcn_wmma_f32_16x16x32_f16(false, za0, false, bw0, (short)0, acc, false, false);
      acc = __builtin_amdgcn_wmma_f32_16x16x32_f16(false, za1, false, bw1, (short)0, acc, false, false);
#pragma unroll
      for (int r = 0; r < 8; ++r)
        h1[(mb + r)*WTS + n] = (_Float16)fmaxf(acc[r], 0.0f);
    }
    __syncthreads();
    // hidden A fragments
    v16h ha0 = SHUF16(*(const v8h*)&h1[m*WTS + loff],
                      *(const v8h*)&h1[m*WTS + loff + 16]);
    v16h ha1 = SHUF16(*(const v8h*)&h1[m*WTS + 32 + loff],
                      *(const v8h*)&h1[m*WTS + 32 + loff + 16]);
    // ---- layer 2 ----
#pragma unroll
    for (int nt = 0; nt < 4; ++nt) {
      int n = (nt << 4) + n15;
      v8f acc;
#pragma unroll
      for (int r = 0; r < 8; ++r) acc[r] = sb2[n];
      v16h bw0 = SHUF16(*(const v8h*)&sw2T[n*WTS + loff],
                        *(const v8h*)&sw2T[n*WTS + loff + 16]);
      v16h bw1 = SHUF16(*(const v8h*)&sw2T[n*WTS + 32 + loff],
                        *(const v8h*)&sw2T[n*WTS + 32 + loff + 16]);
      acc = __builtin_amdgcn_wmma_f32_16x16x32_f16(false, ha0, false, bw0, (short)0, acc, false, false);
      acc = __builtin_amdgcn_wmma_f32_16x16x32_f16(false, ha1, false, bw1, (short)0, acc, false, false);
#pragma unroll
      for (int r = 0; r < 8; ++r)
        out[(size_t)(row0 + mb + r)*DD + n] = acc[r];
    }
    __syncthreads();
  }
}

// ---------------------------------------------------------------------------
extern "C" void kernel_launch(void* const* d_in, const int* in_sizes, int n_in,
                              void* d_out, int out_size, void* d_ws, size_t ws_size,
                              hipStream_t stream) {
  const float* timesteps = (const float*)d_in[0];
  const float* X         = (const float*)d_in[1];
  const float* M         = (const float*)d_in[2];
  const float* yts       = (const float*)d_in[3];
  const float* te_w      = (const float*)d_in[4];
  const float* te_b      = (const float*)d_in[5];
  const float* query     = (const float*)d_in[6];
  const float* qm_w      = (const float*)d_in[7];
  const float* qm_b      = (const float*)d_in[8];
  const float* km_w      = (const float*)d_in[9];
  const float* km_b      = (const float*)d_in[10];
  const float* out_w     = (const float*)d_in[11];
  const float* out_b     = (const float*)d_in[12];
  const float* dec_w1    = (const float*)d_in[13];
  const float* dec_b1    = (const float*)d_in[14];
  const float* dec_w2    = (const float*)d_in[15];
  const float* dec_b2    = (const float*)d_in[16];

  float* ws     = (float*)d_ws;               // needs ~2.1 MB
  float* A      = ws + WS_A;
  float* cvec   = ws + WS_CC;
  float* scores = ws + WS_SCORES;             // becomes expw in place
  float* xatt   = ws + WS_XATT;               // B*SPLIT*4*128 partials

  float* outp   = (float*)d_out;              // (B,T,D) flat
  float* coeffs = outp + OUT0;                // (B,P,LAT) flat, 2nd tuple elem

  k_prep   <<<1,           128, 0, stream>>>(query, qm_w, qm_b, km_w, km_b, A, cvec);
  k_scores <<<(BB*LL)/256, 256, 0, stream>>>(timesteps, te_w, te_b, A, cvec, scores);
  k_expw   <<<BB*4,        256, 0, stream>>>(scores);
  k_attn   <<<BB*SPLIT,    256, 0, stream>>>(X, M, scores, xatt);
  k_coeffs <<<BB,          128, 0, stream>>>(xatt, out_w, out_b, coeffs);
  k_decoder<<<32,          256, 0, stream>>>(yts, coeffs, dec_w1, dec_b1, dec_w2, dec_b2, outp);
}
